// LSTMModel_62852551409927
// MI455X (gfx1250) — matrix operand order repaired and checked by reading, not compiled, not run on hardware
//
#include <hip/hip_runtime.h>
#include <hip/hip_bf16.h>
#include <stddef.h>

// ---------------- problem constants ----------------
#define BATCH   256
#define SEQ     512
#define N_IN    512
#define N_HID   128
#define N_G     512   // 4 * N_HID
#define N_OUT   2

typedef __bf16 bf16_t;
typedef __attribute__((ext_vector_type(16))) __bf16 bf16x16;
typedef __attribute__((ext_vector_type(8)))  float  v8f;

// ---------------- WMMA helpers ----------------
__device__ __forceinline__ v8f wmma_bf16(bf16x16 a, bf16x16 b, v8f c) {
    // (neg_a, A, neg_b, B, c_mod, C, reuse_a, reuse_b)
    return __builtin_amdgcn_wmma_f32_16x16x32_bf16(
        false, a, false, b, (short)0, c, false, false);
}

// A-matrix 16x32 bf16 fragment (row-major source, stride ld elements).
// ISA layout: lanes 0-15 -> M=0..15 ; VGPR0-3: K=(0..7)+8*laneHi ; VGPR4-7: K=16+(0..7)+8*laneHi
__device__ __forceinline__ bf16x16 load_A_frag(const bf16_t* src, int ld, int lane) {
    const int m   = lane & 15;
    const int klo = (lane >> 4) * 8;
    bf16x16 a;
#pragma unroll
    for (int j = 0; j < 8; ++j) a[j]     = src[m * ld + klo + j];
#pragma unroll
    for (int j = 0; j < 8; ++j) a[j + 8] = src[m * ld + 16 + klo + j];
    return a;
}

// fast activations: v_exp_f32 + v_rcp_f32 (serial scan is latency-bound)
__device__ __forceinline__ float fast_sig(float x) {
    return __builtin_amdgcn_rcpf(1.0f + __expf(-x));
}
__device__ __forceinline__ float fast_tanh(float x) {
    return 2.0f * __builtin_amdgcn_rcpf(1.0f + __expf(-2.0f * x)) - 1.0f;
}

// ---------------- kernel 1: pack fp32 weight (K x N, row-major) into ----------------
// WMMA B-fragment order: frag = kt*nFragsN + nt ; per frag 32 lanes x 16 bf16.
// B layout: lane -> N=n ; elems 0-7: K=kt*32+8*laneHi+j ; elems 8-15: K=kt*32+16+8*laneHi+(j-8)
__global__ void pack_wmat_kernel(const float* __restrict__ W, bf16_t* __restrict__ P,
                                 int Ntot, int nFragsN, int totalGid) {
    int gid = blockIdx.x * blockDim.x + threadIdx.x;  // frag*32 + lane
    if (gid >= totalGid) return;
    const int lane = gid & 31;
    const int frag = gid >> 5;
    const int nt   = frag % nFragsN;
    const int kt   = frag / nFragsN;
    const int n    = nt * 16 + (lane & 15);
    const int kb   = kt * 32 + (lane >> 4) * 8;
    bf16x16 t;
#pragma unroll
    for (int j = 0; j < 16; ++j)
        t[j] = (bf16_t)W[(size_t)(kb + j + (j >= 8 ? 8 : 0)) * Ntot + n];
    *(bf16x16*)(P + (size_t)gid * 16) = t;
}

// ---------------- kernel 2: xg[s*B+b][g] = x[b][s][:] @ Wi ----------------
// Block = 256 threads (8 waves). Block tile: M=64 rows of (S*B), N = full 512.
// Wave w: M-subtile (w&3), N-subtiles (w>>2)*16 .. +15.  K loop: 16 iters of 32.
// Per K step: stage x tile (fp32->bf16) AND the frag-contiguous 32 KB Wi kt-slice
// into LDS, then all WMMA operands come from ds_load_b128 (no per-WMMA L2 stalls).
__global__ void __launch_bounds__(256)
xgates_gemm_kernel(const float* __restrict__ x, const bf16_t* __restrict__ WiP,
                   float* __restrict__ xg) {
    __shared__ bf16_t As[64 * 32];        // 4 KB  staged x tile (bf16)
    __shared__ bf16_t Bs[32 * 32 * 16];   // 32 KB staged Wi kt-slice (frag order)

    const int tid  = threadIdx.x;
    const int wave = tid >> 5;
    const int lane = tid & 31;
    const int m0 = blockIdx.x * 64;      // row of (S*B); 64 | 256 => single s per tile
    const int s0 = m0 >> 8;
    const int b0 = m0 & 255;
    const int mt = wave & 3;             // M subtile 0..3
    const int nh = wave >> 2;            // N half 0..1

    v8f acc[16];
#pragma unroll
    for (int j = 0; j < 16; ++j) acc[j] = (v8f){};

    for (int kt = 0; kt < 16; ++kt) {
        // ---- stage 64x32 fp32 -> bf16 x tile into LDS (2 float4 per thread) ----
#pragma unroll
        for (int i = 0; i < 2; ++i) {
            int idx = tid + i * 256;            // 0..511
            int row = idx >> 3;
            int kk  = (idx & 7) * 4;
            const float4 v = *(const float4*)(x + ((size_t)(b0 + row) * SEQ + s0) * N_IN
                                                + kt * 32 + kk);
            As[row * 32 + kk + 0] = (bf16_t)v.x;
            As[row * 32 + kk + 1] = (bf16_t)v.y;
            As[row * 32 + kk + 2] = (bf16_t)v.z;
            As[row * 32 + kk + 3] = (bf16_t)v.w;
        }
        // ---- stage Wi kt-slice: 32 KB contiguous, 8 x uint4 per thread ----
        {
            const uint4* srcv = (const uint4*)WiP + (size_t)kt * 2048;
            uint4* dstv = (uint4*)Bs;
#pragma unroll
            for (int i = 0; i < 8; ++i)
                dstv[tid + i * 256] = srcv[tid + i * 256];
        }
        __syncthreads();

        bf16x16 a = load_A_frag(&As[mt * 16 * 32], 32, lane);
#pragma unroll
        for (int j = 0; j < 16; ++j) {
            int nt = nh * 16 + j;
            bf16x16 bfr = *(const bf16x16*)(Bs + ((size_t)nt * 32 + lane) * 16);
            acc[j] = wmma_bf16(a, bfr, acc[j]);
        }
        __syncthreads();
    }

    // epilogue: D layout VGPR r -> row r + 8*laneHi, col = lane&15
    const int ln = lane & 15;
    const int lh = lane >> 4;
#pragma unroll
    for (int j = 0; j < 16; ++j) {
        int n = (nh * 16 + j) * 16 + ln;
#pragma unroll
        for (int r = 0; r < 8; ++r) {
            int m = mt * 16 + r + lh * 8;
            xg[(size_t)(m0 + m) * N_G + n] = acc[j][r];
        }
    }
}

// ---------------- kernel 3: sequential LSTM scan ----------------
// 16 blocks, each owns a 16-row batch chunk. 256 threads = 8 waves.
// Gate-aligned N mapping: wave w owns nt = {w, 8+w, 16+w, 24+w}, i.e. gate j's
// columns j*128 + [w*16, w*16+16). After WMMA each wave holds i/f/g/o for its
// own (row, col) set => entire LSTM cell computed IN REGISTERS (c also register-
// resident). Only h goes through LDS (double-buffered bf16) => 1 barrier/step.
// Wh fragments (16 per wave) are register-resident for all 512 steps.
__global__ void __launch_bounds__(256)
lstm_scan_kernel(const float* __restrict__ xg, const bf16_t* __restrict__ WhP,
                 const float* __restrict__ bias, float* __restrict__ hLast) {
    __shared__ bf16_t hbuf[2][16 * N_HID];   // 2 x 4 KB, double-buffered h state

    const int tid  = threadIdx.x;
    const int wave = tid >> 5;
    const int lane = tid & 31;
    const int b0   = blockIdx.x * 16;     // batch chunk base
    const int ln   = lane & 15;
    const int lh   = lane >> 4;
    const int hcol = wave * 16 + ln;      // this lane's hidden column (0..127)

    for (int i = tid; i < 16 * N_HID; i += 256)
        hbuf[0][i] = (bf16_t)0.0f;

    // ---- per-lane bias for the 4 gates (column is fixed per lane) ----
    float biasv[4];
#pragma unroll
    for (int j = 0; j < 4; ++j) biasv[j] = bias[j * 128 + hcol];

    // ---- this wave's 16 Wh fragments -> registers (persist all steps) ----
    bf16x16 whf[4][4];   // [kt][gate j], nt = j*8 + wave
#pragma unroll
    for (int kt = 0; kt < 4; ++kt)
#pragma unroll
        for (int j = 0; j < 4; ++j)
            whf[kt][j] = *(const bf16x16*)(WhP +
                ((size_t)(kt * 32 + j * 8 + wave) * 32 + lane) * 16);

    // ---- register-resident cell state: 8 rows per lane ----
    float creg[8];
#pragma unroll
    for (int r = 0; r < 8; ++r) creg[r] = 0.0f;

    // ---- xg for step 0 ----
    float xv[4][8];
#pragma unroll
    for (int j = 0; j < 4; ++j)
#pragma unroll
        for (int r = 0; r < 8; ++r)
            xv[j][r] = xg[(size_t)(b0 + r + lh * 8) * N_G + j * 128 + hcol];

    __syncthreads();

    for (int s = 0; s < SEQ; ++s) {
        const bf16_t* hb_cur = hbuf[s & 1];
        bf16_t*       hb_nxt = hbuf[(s + 1) & 1];

        // ---- warm L2 for step s+2's xg tile (32 KB = 128 B/thread) ----
        if (s + 2 < SEQ) {
            const char* nxt = (const char*)(xg + (size_t)((s + 2) * BATCH + b0) * N_G);
            __builtin_prefetch(nxt + tid * 128, 0, 1);
        }

        // ---- 16 WMMAs; A-frags pipelined one kt ahead, Wh in registers ----
        v8f acc[4];
#pragma unroll
        for (int j = 0; j < 4; ++j) acc[j] = (v8f){};

        bf16x16 a = load_A_frag(hb_cur, N_HID, lane);
#pragma unroll
        for (int kt = 0; kt < 4; ++kt) {
            bf16x16 anext;
            if (kt < 3) anext = load_A_frag(hb_cur + (kt + 1) * 32, N_HID, lane);
#pragma unroll
            for (int j = 0; j < 4; ++j)
                acc[j] = wmma_bf16(a, whf[kt][j], acc[j]);
            if (kt < 3) a = anext;
        }

        // ---- fused LSTM cell entirely in registers ----
        float hnew[8];
#pragma unroll
        for (int r = 0; r < 8; ++r) {
            float gi = acc[0][r] + xv[0][r] + biasv[0];
            float gf = acc[1][r] + xv[1][r] + biasv[1];
            float gg = acc[2][r] + xv[2][r] + biasv[2];
            float go = acc[3][r] + xv[3][r] + biasv[3];
            float c  = fast_sig(gf) * creg[r] + fast_sig(gi) * fast_tanh(gg);
            creg[r]  = c;
            hnew[r]  = fast_sig(go) * fast_tanh(c);
        }

        // ---- issue NEXT step's xg loads; overlap h-store + barrier + A-frags ----
        if (s + 1 < SEQ) {
            const float* xgrow = xg + (size_t)((s + 1) * BATCH + b0) * N_G;
#pragma unroll
            for (int j = 0; j < 4; ++j)
#pragma unroll
                for (int r = 0; r < 8; ++r)
                    xv[j][r] = xgrow[(size_t)(r + lh * 8) * N_G + j * 128 + hcol];
        }

        // ---- publish h (bf16) to the other buffer ----
#pragma unroll
        for (int r = 0; r < 8; ++r)
            hb_nxt[(r + lh * 8) * N_HID + hcol] = (bf16_t)hnew[r];

        __syncthreads();
    }

    // ---- epilogue: final h lives in hbuf[0] (SEQ even) ----
    for (int i = tid; i < 16 * N_HID; i += 256) {
        int bb = i >> 7, hh = i & 127;
        hLast[(size_t)(b0 + bb) * N_HID + hh] = (float)hbuf[0][i];
    }
}

// ---------------- kernel 4: logits = h_last @ Wd + bd ----------------
__global__ void dense_out_kernel(const float* __restrict__ hLast,
                                 const float* __restrict__ Wd,
                                 const float* __restrict__ bd,
                                 float* __restrict__ out) {
    int tid = threadIdx.x;           // 0..511
    int b = tid >> 1, o = tid & 1;
    float s = bd[o];
#pragma unroll 8
    for (int k = 0; k < N_HID; ++k)
        s += hLast[b * N_HID + k] * Wd[k * N_OUT + o];
    out[b * N_OUT + o] = s;
}

// ---------------- host launcher ----------------
extern "C" void kernel_launch(void* const* d_in, const int* in_sizes, int n_in,
                              void* d_out, int out_size, void* d_ws, size_t ws_size,
                              hipStream_t stream) {
    (void)in_sizes; (void)n_in; (void)out_size; (void)ws_size;
    const float* x  = (const float*)d_in[0];
    const float* Wi = (const float*)d_in[1];
    const float* Wh = (const float*)d_in[2];
    const float* b  = (const float*)d_in[3];
    const float* Wd = (const float*)d_in[4];
    const float* bd = (const float*)d_in[5];
    float* out = (float*)d_out;

    // workspace layout
    const size_t xg_bytes  = (size_t)SEQ * BATCH * N_G * sizeof(float);   // 268 MB
    const size_t wip_bytes = (size_t)N_IN  * N_G * sizeof(bf16_t);        // 512 KB
    const size_t whp_bytes = (size_t)N_HID * N_G * sizeof(bf16_t);        // 128 KB
    char* ws = (char*)d_ws;
    float*  xg    = (float*)ws;
    bf16_t* WiP   = (bf16_t*)(ws + xg_bytes);
    bf16_t* WhP   = (bf16_t*)(ws + xg_bytes + wip_bytes);
    float*  hLast = (float*)(ws + xg_bytes + wip_bytes + whp_bytes);

    // pack Wi: K=512 -> 16 kt, N=512 -> 32 nt => 512 frags * 32 lanes = 16384 gids
    pack_wmat_kernel<<<64, 256, 0, stream>>>(Wi, WiP, N_G, 32, 16384);
    // pack Wh: K=128 -> 4 kt, N=512 -> 32 nt => 128 frags * 32 lanes = 4096 gids
    pack_wmat_kernel<<<16, 256, 0, stream>>>(Wh, WhP, N_G, 32, 4096);

    // x_gates GEMM: (S*B)/64 = 2048 blocks
    xgates_gemm_kernel<<<(SEQ * BATCH) / 64, 256, 0, stream>>>(x, WiP, xg);

    // recurrent scan: 16 batch chunks
    lstm_scan_kernel<<<BATCH / 16, 256, 0, stream>>>(xg, WhP, b, hLast);

    // final dense layer
    dense_out_kernel<<<1, BATCH * N_OUT, 0, stream>>>(hLast, Wd, bd, out);
}
